// LSTM_45578192945177
// MI455X (gfx1250) — compile-verified, matrix operand order checked by reading, not run on hardware
//
#include <hip/hip_runtime.h>
#include <hip/hip_bf16.h>
#include <math.h>

// LSTM cell: B=1024, H=2048, E=1024, fp32 in/out, bf16 WMMA compute.
#define BB 1024
#define HH 2048
#define EE 1024

typedef __attribute__((ext_vector_type(16))) __bf16 v16bf;
typedef __attribute__((ext_vector_type(8)))  float  v8f;
typedef __attribute__((ext_vector_type(4)))  float  f32x4;

// Raw fp32 data for one 16x32 bf16 WMMA fragment (this lane's share):
// two contiguous 32B chunks at +0 and +16 floats (ISA 7.12.2 half-wave split).
struct rawfrag { f32x4 c0, c1, c2, c3; };

__device__ __forceinline__ rawfrag load_raw(const float* p) {
    const f32x4* q = (const f32x4*)p;
    rawfrag r;
    r.c0 = q[0];   // k+0..3
    r.c1 = q[1];   // k+4..7
    r.c2 = q[4];   // k+16..19
    r.c3 = q[5];   // k+20..23
    return r;
}

__device__ __forceinline__ v16bf cvt_frag(const rawfrag& f) {
    v16bf r;
#pragma unroll
    for (int i = 0; i < 4; ++i) {
        r[i]      = (__bf16)f.c0[i];
        r[4 + i]  = (__bf16)f.c1[i];
        r[8 + i]  = (__bf16)f.c2[i];
        r[12 + i] = (__bf16)f.c3[i];
    }
    return r;
}

// reuse-A operand-cache hint must be an immediate -> template parameter.
template <bool REUSE_A>
__device__ __forceinline__ v8f wmma_bf16(v16bf a, v16bf b, v8f acc) {
    return __builtin_amdgcn_wmma_f32_16x16x32_bf16(
        /*neg_a=*/false, a, /*neg_b=*/false, b,
        /*c_mod=*/(short)0, acc, /*reuse_a=*/REUSE_A, /*reuse_b=*/false);
}

// One GEMM phase: acc{0..3} += A(16xK) * W{0..3}(16xK)^T, software-pipelined
// one K-step ahead so next-step loads are in flight during cvt+wmma.
__device__ __forceinline__ void gemm_phase4(
    const float* __restrict__ ax,
    const float* __restrict__ w0, const float* __restrict__ w1,
    const float* __restrict__ w2, const float* __restrict__ w3,
    int K, v8f& acc0, v8f& acc1, v8f& acc2, v8f& acc3)
{
    rawfrag ra = load_raw(ax);
    rawfrag r0 = load_raw(w0);
    rawfrag r1 = load_raw(w1);
    rawfrag r2 = load_raw(w2);
    rawfrag r3 = load_raw(w3);
    for (int kb = 32; kb < K; kb += 32) {
        // issue next K-step loads before touching current data
        rawfrag na = load_raw(ax + kb);
        rawfrag n0 = load_raw(w0 + kb);
        rawfrag n1 = load_raw(w1 + kb);
        rawfrag n2 = load_raw(w2 + kb);
        rawfrag n3 = load_raw(w3 + kb);
        v16bf a = cvt_frag(ra);
        acc0 = wmma_bf16<true >(a, cvt_frag(r0), acc0);
        acc1 = wmma_bf16<true >(a, cvt_frag(r1), acc1);
        acc2 = wmma_bf16<true >(a, cvt_frag(r2), acc2);
        acc3 = wmma_bf16<false>(a, cvt_frag(r3), acc3);
        ra = na; r0 = n0; r1 = n1; r2 = n2; r3 = n3;
    }
    v16bf a = cvt_frag(ra);
    acc0 = wmma_bf16<true >(a, cvt_frag(r0), acc0);
    acc1 = wmma_bf16<true >(a, cvt_frag(r1), acc1);
    acc2 = wmma_bf16<true >(a, cvt_frag(r2), acc2);
    acc3 = wmma_bf16<false>(a, cvt_frag(r3), acc3);
}

__device__ __forceinline__ float sigmoidf_(float z) {
    return __builtin_amdgcn_rcpf(1.0f + __expf(-z));
}
__device__ __forceinline__ float tanhf_(float z) {
    // tanh(z) = 1 - 2/(exp(2z)+1)
    return 1.0f - 2.0f * __builtin_amdgcn_rcpf(__expf(z + z) + 1.0f);
}

// ---------------------------------------------------------------------------
// Kernel 1: all 4 gates fused. Each wave computes one 16(batch) x 16(hidden)
// tile for i/g/f/o, then c_new = f*c + i*g, h_new = o*c_new in registers.
// Grid: (B/16)*(H/16) = 8192 waves -> 1024 blocks of 8 waves.
// ---------------------------------------------------------------------------
__global__ __launch_bounds__(256)
void lstm_gates_wmma(const float* __restrict__ x,
                     const float* __restrict__ c,
                     const float* __restrict__ h,
                     const float* __restrict__ Wxi, const float* __restrict__ Whi, const float* __restrict__ Bi,
                     const float* __restrict__ Wxg, const float* __restrict__ Whg, const float* __restrict__ Bg,
                     const float* __restrict__ Wxf, const float* __restrict__ Whf, const float* __restrict__ Bf,
                     const float* __restrict__ Wxo, const float* __restrict__ Who, const float* __restrict__ Bo,
                     float* __restrict__ c_out, float* __restrict__ h_out)
{
    const int lane   = threadIdx.x & 31;
    const int wave   = threadIdx.x >> 5;
    const int wtile  = blockIdx.x * 8 + wave;
    const int ntiles = HH / 16;                 // 128
    const int m0 = (wtile / ntiles) * 16;       // batch-row base
    const int n0 = (wtile % ntiles) * 16;       // hidden-col base

    const int lrow = lane & 15;                 // A row / B column within tile
    const int ksel = (lane >> 4) * 8;           // half-wave K select

    v8f acc_i = {}; v8f acc_g = {}; v8f acc_f = {}; v8f acc_o = {};

    // ---- X * Wx^T, K = E -------------------------------------------------
    gemm_phase4(x   + (size_t)(m0 + lrow) * EE + ksel,
                Wxi + (size_t)(n0 + lrow) * EE + ksel,
                Wxg + (size_t)(n0 + lrow) * EE + ksel,
                Wxf + (size_t)(n0 + lrow) * EE + ksel,
                Wxo + (size_t)(n0 + lrow) * EE + ksel,
                EE, acc_i, acc_g, acc_f, acc_o);

    // ---- H * Wh^T, K = H -------------------------------------------------
    gemm_phase4(h   + (size_t)(m0 + lrow) * HH + ksel,
                Whi + (size_t)(n0 + lrow) * HH + ksel,
                Whg + (size_t)(n0 + lrow) * HH + ksel,
                Whf + (size_t)(n0 + lrow) * HH + ksel,
                Who + (size_t)(n0 + lrow) * HH + ksel,
                HH, acc_i, acc_g, acc_f, acc_o);

    // ---- fused gate epilogue --------------------------------------------
    // C/D layout: lane L holds col n = L&15; elems r map to rows m0+(L<16?0:8)+r.
    const int   n  = n0 + lrow;
    const int   mb = m0 + ((lane >> 4) * 8);
    const float bi = Bi[n], bg = Bg[n], bf = Bf[n], bo = Bo[n];
#pragma unroll
    for (int r = 0; r < 8; ++r) {
        const size_t idx = (size_t)(mb + r) * HH + n;
        const float iv = sigmoidf_(acc_i[r] + bi);
        const float gv = tanhf_   (acc_g[r] + bg);
        const float fv = sigmoidf_(acc_f[r] + bf);
        const float ov = sigmoidf_(acc_o[r] + bo);
        const float cn = fv * c[idx] + iv * gv;
        c_out[idx] = cn;
        h_out[idx] = ov * cn;   // faithful to reference: o * c_new, no tanh
    }
}

// ---------------------------------------------------------------------------
// Kernel 2: y = tanh(h_new * Why^T + By). Each wave: 16(batch) x 64(out) via
// 4 N-subtiles sharing one A fragment per K-step (K = H).
// Grid: (B/16)*(E/64) = 1024 waves -> 128 blocks of 8 waves.
// ---------------------------------------------------------------------------
__global__ __launch_bounds__(256)
void lstm_out_wmma(const float* __restrict__ hnew,
                   const float* __restrict__ Why, const float* __restrict__ By,
                   float* __restrict__ y)
{
    const int lane   = threadIdx.x & 31;
    const int wave   = threadIdx.x >> 5;
    const int wtile  = blockIdx.x * 8 + wave;
    const int ntiles = EE / 64;                 // 16
    const int m0 = (wtile / ntiles) * 16;
    const int n0 = (wtile % ntiles) * 64;

    const int lrow = lane & 15;
    const int ksel = (lane >> 4) * 8;

    v8f acc0 = {}; v8f acc1 = {}; v8f acc2 = {}; v8f acc3 = {};

    gemm_phase4(hnew + (size_t)(m0 + lrow) * HH + ksel,
                Why + (size_t)(n0 +  0 + lrow) * HH + ksel,
                Why + (size_t)(n0 + 16 + lrow) * HH + ksel,
                Why + (size_t)(n0 + 32 + lrow) * HH + ksel,
                Why + (size_t)(n0 + 48 + lrow) * HH + ksel,
                HH, acc0, acc1, acc2, acc3);

    const int mb = m0 + ((lane >> 4) * 8);
    v8f accs[4] = {acc0, acc1, acc2, acc3};
#pragma unroll
    for (int t = 0; t < 4; ++t) {
        const int   n  = n0 + t * 16 + lrow;
        const float by = By[n];
#pragma unroll
        for (int r = 0; r < 8; ++r) {
            y[(size_t)(mb + r) * EE + n] = tanhf_(accs[t][r] + by);
        }
    }
}

// ---------------------------------------------------------------------------
extern "C" void kernel_launch(void* const* d_in, const int* in_sizes, int n_in,
                              void* d_out, int out_size, void* d_ws, size_t ws_size,
                              hipStream_t stream) {
    const float* x   = (const float*)d_in[0];
    const float* c   = (const float*)d_in[1];
    const float* h   = (const float*)d_in[2];
    const float* Wxi = (const float*)d_in[3];
    const float* Whi = (const float*)d_in[4];
    const float* Bi  = (const float*)d_in[5];
    const float* Wxg = (const float*)d_in[6];
    const float* Whg = (const float*)d_in[7];
    const float* Bg  = (const float*)d_in[8];
    const float* Wxf = (const float*)d_in[9];
    const float* Whf = (const float*)d_in[10];
    const float* Bf  = (const float*)d_in[11];
    const float* Wxo = (const float*)d_in[12];
    const float* Who = (const float*)d_in[13];
    const float* Bo  = (const float*)d_in[14];
    const float* Why = (const float*)d_in[15];
    const float* By  = (const float*)d_in[16];

    // d_out = [ y (B*E) | c_new (B*H) | h_new (B*H) ]
    float* y_out = (float*)d_out;
    float* c_out = y_out + (size_t)BB * EE;
    float* h_out = c_out + (size_t)BB * HH;

    // Gates: (B/16)*(H/16) = 8192 wave-tiles / 8 waves per block
    lstm_gates_wmma<<<(BB / 16) * (HH / 16) / 8, 256, 0, stream>>>(
        x, c, h,
        Wxi, Whi, Bi, Wxg, Whg, Bg, Wxf, Whf, Bf, Wxo, Who, Bo,
        c_out, h_out);

    // Output projection: (B/16)*(E/64) = 1024 wave-tiles / 8 waves per block
    lstm_out_wmma<<<(BB / 16) * (EE / 64) / 8, 256, 0, stream>>>(
        h_out, Why, By, y_out);
}
